// RotaryPositionAttention_62457414419232
// MI455X (gfx1250) — matrix operand order repaired and checked by reading, not compiled, not run on hardware
//
#include <hip/hip_runtime.h>
#include <hip/hip_bf16.h>

// ---------------- problem constants (match reference) ----------------
#define B_   4
#define T_   2048
#define DIM_ 1024
#define H_   16
#define HD_  64
#define ND3_ 3072   // 3*DIM

typedef __attribute__((ext_vector_type(16))) __bf16         v16bf;
typedef __attribute__((ext_vector_type(8)))  float          v8f;
typedef __attribute__((ext_vector_type(8)))  unsigned short ushort8;
typedef __attribute__((ext_vector_type(4)))  unsigned int   u32x4;
typedef __attribute__((ext_vector_type(4)))  int            i32x4;
typedef __attribute__((ext_vector_type(8)))  int            i32x8;

// f32 -> bf16 with round-to-nearest-even
static __device__ __forceinline__ unsigned short f2bf(float f) {
  unsigned int u = __float_as_uint(f);
  unsigned int r = (u + 0x7FFFu + ((u >> 16) & 1u)) >> 16;
  return (unsigned short)r;
}

// ---- WMMA fragment loads (ISA 7.12.2 layouts, wave32) ----
// A (16x32 bf16): lane m = l&15; lanes<16 hold K {0..7,16..23}, lanes>=16 hold K {8..15,24..31}
static __device__ __forceinline__ v16bf load_frag_a(const unsigned short* base, int ld, int lane) {
  const int m  = lane & 15;
  const int kh = (lane >> 4) << 3;  // 0 or 8
  union { v16bf v; ushort8 h[2]; } u;
  u.h[0] = *(const ushort8*)(base + m * ld + kh);
  u.h[1] = *(const ushort8*)(base + m * ld + 16 + kh);
  return u.v;
}
// B (32x16 bf16), source stored as rows of N with K contiguous (NT layout):
// lane n = l&15; lanes<16 hold K {0..7,16..23}, lanes>=16 hold K {8..15,24..31}
static __device__ __forceinline__ v16bf load_frag_b(const unsigned short* base, int ld, int lane) {
  const int n  = lane & 15;
  const int ko = (lane >> 4) << 3;  // 0 or 8
  union { v16bf v; ushort8 h[2]; } u;
  u.h[0] = *(const ushort8*)(base + n * ld + ko);
  u.h[1] = *(const ushort8*)(base + n * ld + 16 + ko);
  return u.v;
}

static __device__ __forceinline__ v8f wmma_bf16(v16bf a, v16bf b, v8f c) {
  return __builtin_amdgcn_wmma_f32_16x16x32_bf16(false, a, false, b, (short)0, c, false, false);
}

static __device__ __forceinline__ float rmax16(float x) {
  #pragma unroll
  for (int m = 1; m < 16; m <<= 1) x = fmaxf(x, __shfl_xor(x, m, 32));
  return x;
}
static __device__ __forceinline__ float rsum16(float x) {
  #pragma unroll
  for (int m = 1; m < 16; m <<= 1) x += __shfl_xor(x, m, 32);
  return x;
}

// =====================================================================
// GEMM (NT): C[M,N] = A[M,K] * W[N,K]^T (+bias), f32 in, bf16 WMMA, f32 out
// block tile 128(M) x 128(N), BK=32; 256 threads = 8 waves, each wave 32x64
// => 8 WMMAs per wave per K-step on 2 A-frags x 4 B-frags
// =====================================================================
__global__ __launch_bounds__(256)
void gemm_nt_bf16(const float* __restrict__ A, const float* __restrict__ W,
                  float* __restrict__ C, const float* __restrict__ bias,
                  int M, int N, int K) {
  __shared__ unsigned short As[128][40];   // 128 x 32 (+8 pad), bf16
  __shared__ unsigned short Bs[128][40];   // 128 x 32 (+8 pad), bf16

  const int tid  = threadIdx.x;
  const int lane = tid & 31;
  const int wave = tid >> 5;
  const int m0   = (wave & 3) * 32;   // wave M sub-tile inside block
  const int n0w  = (wave >> 2) * 64;  // wave N sub-tile inside block
  const int mb   = blockIdx.y * 128;
  const int nb   = blockIdx.x * 128;

  const v8f zero = {0.f, 0.f, 0.f, 0.f, 0.f, 0.f, 0.f, 0.f};
  v8f acc[2][4];
  #pragma unroll
  for (int i = 0; i < 2; ++i)
    #pragma unroll
    for (int j = 0; j < 4; ++j) acc[i][j] = zero;

  const int srow = tid >> 1, scol = (tid & 1) * 16;  // 2 thr/row, 16 f32 each

  for (int kt = 0; kt < K; kt += 32) {
    { // stage A tile 128x32 (f32 -> bf16)
      const float* src = A + (size_t)(mb + srow) * K + kt + scol;
      #pragma unroll
      for (int q = 0; q < 2; ++q) {
        float4 f0 = *(const float4*)(src + q * 8);
        float4 f1 = *(const float4*)(src + q * 8 + 4);
        ushort8 h;
        h[0] = f2bf(f0.x); h[1] = f2bf(f0.y); h[2] = f2bf(f0.z); h[3] = f2bf(f0.w);
        h[4] = f2bf(f1.x); h[5] = f2bf(f1.y); h[6] = f2bf(f1.z); h[7] = f2bf(f1.w);
        *(ushort8*)&As[srow][scol + q * 8] = h;
      }
    }
    { // stage B tile 128x32
      const float* src = W + (size_t)(nb + srow) * K + kt + scol;
      #pragma unroll
      for (int q = 0; q < 2; ++q) {
        float4 f0 = *(const float4*)(src + q * 8);
        float4 f1 = *(const float4*)(src + q * 8 + 4);
        ushort8 h;
        h[0] = f2bf(f0.x); h[1] = f2bf(f0.y); h[2] = f2bf(f0.z); h[3] = f2bf(f0.w);
        h[4] = f2bf(f1.x); h[5] = f2bf(f1.y); h[6] = f2bf(f1.z); h[7] = f2bf(f1.w);
        *(ushort8*)&Bs[srow][scol + q * 8] = h;
      }
    }
    __syncthreads();

    v16bf a0 = load_frag_a(&As[m0][0], 40, lane);
    v16bf a1 = load_frag_a(&As[m0 + 16][0], 40, lane);
    #pragma unroll
    for (int j = 0; j < 4; ++j) {
      v16bf b = load_frag_b(&Bs[n0w + j * 16][0], 40, lane);
      acc[0][j] = wmma_bf16(a0, b, acc[0][j]);
      acc[1][j] = wmma_bf16(a1, b, acc[1][j]);
    }
    __syncthreads();
  }

  // epilogue: C fragment layout (lane n = l&15; rows r + 8*(l>=16))
  const int n = lane & 15;
  const int mbase = (lane >> 4) * 8;
  #pragma unroll
  for (int j = 0; j < 4; ++j) {
    const int gn = nb + n0w + j * 16 + n;
    const float bv = bias ? bias[gn] : 0.f;
    #pragma unroll
    for (int i = 0; i < 2; ++i) {
      #pragma unroll
      for (int r = 0; r < 8; ++r) {
        const int gm = mb + m0 + i * 16 + mbase + r;
        C[(size_t)gm * N + gn] = acc[i][j][r] + bv;
      }
    }
  }
}

// =====================================================================
// RoPE (interleaved pairs) + split qkv[B,T,3,H,HD] f32 -> q/k/v bf16 [B,H,T,HD]
// =====================================================================
__global__ __launch_bounds__(256)
void rope_split(const float* __restrict__ qkv,
                unsigned short* __restrict__ qb,
                unsigned short* __restrict__ kb,
                unsigned short* __restrict__ vb) {
  const unsigned int gid = blockIdx.x * 256u + threadIdx.x;
  const int i = gid & 31;            // pair index 0..31
  const int h = (gid >> 5) & 15;
  const int t = (gid >> 9) & (T_ - 1);
  const int b = gid >> 20;

  const float* row = qkv + (size_t)(b * T_ + t) * ND3_;
  const int c0 = h * HD_ + 2 * i;
  const float q0 = row[c0],            q1 = row[c0 + 1];
  const float k0 = row[DIM_ + c0],     k1 = row[DIM_ + c0 + 1];
  const float v0 = row[2 * DIM_ + c0], v1 = row[2 * DIM_ + c0 + 1];

  // inv_freq = 10000^(-2i/64) = exp(-(ln 1e4 / 32) * i)
  const float inv = __expf(-0.2878231366f * (float)i);
  const float ang = (float)t * inv;
  const float c = __cosf(ang), s = __sinf(ang);

  const size_t o = (((size_t)(b * H_ + h)) * T_ + t) * HD_ + 2 * i;
  qb[o]     = f2bf(q0 * c - q1 * s);
  qb[o + 1] = f2bf(q1 * c + q0 * s);
  kb[o]     = f2bf(k0 * c - k1 * s);
  kb[o + 1] = f2bf(k1 * c + k0 * s);
  vb[o]     = f2bf(v0);
  vb[o + 1] = f2bf(v1);
}

// =====================================================================
// TDM staging of the K chunk (bf16 verbatim copy, with D# LDS padding to
// produce the 72-ushort row stride). Guarded: falls back to manual copy.
// =====================================================================
#define TDM_OK __has_builtin(__builtin_amdgcn_tensor_load_to_lds)

#if TDM_OK
static __device__ __forceinline__ void tdm_load_k_chunk(
    const unsigned short* gsrc /* K rows, 32x64 bf16 */, unsigned int lds_byte_addr) {
  const unsigned long long ga = (unsigned long long)(uintptr_t)gsrc;
  u32x4 g0;
  g0[0] = 1u;                                   // count=1, is_restore=0, gather=0
  g0[1] = lds_byte_addr;                        // lds_addr
  g0[2] = (unsigned int)(ga & 0xFFFFFFFFull);   // global_addr[31:0]
  g0[3] = (unsigned int)((ga >> 32) & 0x01FFFFFFull) | (2u << 30);  // addr[56:32], type=2

  i32x8 g1;
  // workgroup_mask=0 | data_size=1(2B) | pad_enable | pad_interval=4(32 dw) | pad_amount=3(4 dw)
  g1[0] = (1 << 16) | (1 << 20) | (4 << 22) | (3 << 25);
  g1[1] = (HD_ & 0xFFFF) << 16;                 // tensor_dim0 = 64 (bits 79:48)
  g1[2] = (T_ & 0xFFFF) << 16;                  // tensor_dim1 = 2048 (bits 111:80)
  g1[3] = (HD_ & 0xFFFF) << 16;                 // tile_dim0 = 64 (bits 127:112)
  g1[4] = 32;                                   // tile_dim1 = 32 keys
  g1[5] = HD_;                                  // tensor_dim0_stride = 64
  g1[6] = 0;
  g1[7] = 0;

  i32x4 gz = {0, 0, 0, 0};
#if __clang_major__ >= 23
  i32x8 gz8 = {0, 0, 0, 0, 0, 0, 0, 0};
  __builtin_amdgcn_tensor_load_to_lds(g0, g1, gz, gz, gz8, 0);
#else
  __builtin_amdgcn_tensor_load_to_lds(g0, g1, gz, gz, 0);
#endif
  __builtin_amdgcn_s_wait_tensorcnt(0);
}
#endif

// =====================================================================
// Flash attention: 8 waves/block, each wave owns a 16-row Q tile (block = 128 q rows).
// K chunk staged via TDM (tensor_load_to_lds) when available; V staged manually
// transposed so P@V B-fragments are contiguous ds loads. Online softmax with
// shfl_xor row reductions.
// =====================================================================
__global__ __launch_bounds__(256)
void flash_attn(const unsigned short* __restrict__ qb,
                const unsigned short* __restrict__ kb,
                const unsigned short* __restrict__ vb,
                float* __restrict__ out) {
  __shared__ unsigned short Ks[32][72];      // [key][hd]   (64+8 pad)
  __shared__ unsigned short Vt[64][40];      // [hd][key]   (32+8 pad)
  __shared__ unsigned short Ps[8][16][40];   // per-wave P scratch [m][key]

  const int tid  = threadIdx.x;
  const int lane = tid & 31;
  const int wave = tid >> 5;
  const int nblk = T_ / 128;
  const int bh   = blockIdx.x / nblk;
  const int qblk = blockIdx.x % nblk;
  const int b    = bh >> 4, h = bh & 15;
  const int t0   = qblk * 128 + wave * 16;

  const unsigned short* qrow  = qb + ((size_t)bh * T_ + t0) * HD_;
  const unsigned short* kbase = kb + (size_t)bh * T_ * HD_;
  const unsigned short* vbase = vb + (size_t)bh * T_ * HD_;

  v16bf aq0 = load_frag_a(qrow, HD_, lane);        // hd 0..31
  v16bf aq1 = load_frag_a(qrow + 32, HD_, lane);   // hd 32..63

  const v8f zero = {0.f, 0.f, 0.f, 0.f, 0.f, 0.f, 0.f, 0.f};
  v8f o_acc[4];
  #pragma unroll
  for (int j = 0; j < 4; ++j) o_acc[j] = zero;
  float run_max[8], run_sum[8];
  #pragma unroll
  for (int r = 0; r < 8; ++r) { run_max[r] = -3.0e38f; run_sum[r] = 0.f; }

  const float scale = 0.125f;  // 1/sqrt(64)
  const int krow = tid >> 3;        // 0..31 (key within chunk)
  const int kcol = (tid & 7) * 8;   // 0..56 (hd)

#if TDM_OK
  const unsigned int ks_lds = (unsigned int)(uintptr_t)&Ks[0][0];
#endif

  for (int kt = 0; kt < T_; kt += 32) {
    // ---- stage K chunk [32 x 64] ----
#if TDM_OK
    if (wave == 0) tdm_load_k_chunk(kbase + (size_t)kt * HD_, ks_lds);
#else
    {
      ushort8 kv = *(const ushort8*)(kbase + (size_t)(kt + krow) * HD_ + kcol);
      *(ushort8*)&Ks[krow][kcol] = kv;
    }
#endif
    // ---- stage V chunk transposed [64 x 32] ----
    {
      ushort8 vv = *(const ushort8*)(vbase + (size_t)(kt + krow) * HD_ + kcol);
      #pragma unroll
      for (int e = 0; e < 8; ++e) Vt[kcol + e][krow] = vv[e];
    }
    if (kt + 32 < T_) {  // global_prefetch_b8 for the next chunk
      __builtin_prefetch(kbase + (size_t)(kt + 32 + krow) * HD_ + kcol, 0, 1);
      __builtin_prefetch(vbase + (size_t)(kt + 32 + krow) * HD_ + kcol, 0, 1);
    }
    __syncthreads();

    // ---- S = Q @ K^T  (16 x 32) ----
    v8f s0 = zero, s1 = zero;
    {
      v16bf b0 = load_frag_b(&Ks[0][0], 72, lane);
      v16bf b1 = load_frag_b(&Ks[0][32], 72, lane);
      s0 = wmma_bf16(aq0, b0, s0);
      s0 = wmma_bf16(aq1, b1, s0);
      v16bf b2 = load_frag_b(&Ks[16][0], 72, lane);
      v16bf b3 = load_frag_b(&Ks[16][32], 72, lane);
      s1 = wmma_bf16(aq0, b2, s1);
      s1 = wmma_bf16(aq1, b3, s1);
    }

    // ---- online softmax (row-wise over keys = lane dimension) ----
    #pragma unroll
    for (int r = 0; r < 8; ++r) {
      const float x0 = s0[r] * scale, x1 = s1[r] * scale;
      float cm = rmax16(fmaxf(x0, x1));
      const float nm   = fmaxf(run_max[r], cm);
      const float corr = __expf(run_max[r] - nm);
      const float p0 = __expf(x0 - nm), p1 = __expf(x1 - nm);
      run_sum[r] = run_sum[r] * corr + rsum16(p0 + p1);
      run_max[r] = nm;
      const int mrow = r + (lane >> 4) * 8;
      Ps[wave][mrow][lane & 15]        = f2bf(p0);
      Ps[wave][mrow][16 + (lane & 15)] = f2bf(p1);
      #pragma unroll
      for (int jh = 0; jh < 4; ++jh) o_acc[jh][r] *= corr;
    }

    // ---- O += P @ V  (contract over 32 keys) ----
    v16bf ap = load_frag_a(&Ps[wave][0][0], 40, lane);
    #pragma unroll
    for (int jh = 0; jh < 4; ++jh) {
      v16bf bvf = load_frag_b(&Vt[jh * 16][0], 40, lane);
      o_acc[jh] = wmma_bf16(ap, bvf, o_acc[jh]);
    }
    __syncthreads();
  }

  // ---- normalize and store: out[B,T, h*64+hd] (fp32, feeds final GEMM) ----
  const int n = lane & 15;
  const int mbase = (lane >> 4) * 8;
  #pragma unroll
  for (int r = 0; r < 8; ++r) {
    const float invs = 1.0f / run_sum[r];
    const int t = t0 + mbase + r;
    float* orow = out + ((size_t)(b * T_ + t)) * DIM_ + h * HD_;
    #pragma unroll
    for (int jh = 0; jh < 4; ++jh) orow[jh * 16 + n] = o_acc[jh][r] * invs;
  }
}

// =====================================================================
extern "C" void kernel_launch(void* const* d_in, const int* in_sizes, int n_in,
                              void* d_out, int out_size, void* d_ws, size_t ws_size,
                              hipStream_t stream) {
  (void)in_sizes; (void)n_in; (void)out_size; (void)ws_size;

  const float* x      = (const float*)d_in[0];
  const float* w_qkv  = (const float*)d_in[1];
  const float* w_proj = (const float*)d_in[2];
  const float* b_proj = (const float*)d_in[3];
  float* out = (float*)d_out;

  char* ws = (char*)d_ws;
  size_t off = 0;
  auto alloc = [&](size_t bytes) -> void* {
    void* p = ws + off;
    off = (off + bytes + 255) & ~(size_t)255;
    return p;
  };

  const size_t MT = (size_t)B_ * T_;  // 8192 rows
  float*          qkv  = (float*)alloc(MT * ND3_ * sizeof(float));   // 96 MB
  unsigned short* qb   = (unsigned short*)alloc(MT * DIM_ * 2);      // 16 MB
  unsigned short* kb2  = (unsigned short*)alloc(MT * DIM_ * 2);      // 16 MB
  unsigned short* vb2  = (unsigned short*)alloc(MT * DIM_ * 2);      // 16 MB
  float*          attn = (float*)alloc(MT * DIM_ * sizeof(float));   // 32 MB

  dim3 blk(256);

  // 1) qkv = x @ w_qkv^T     (M=8192, N=3072, K=1024)
  gemm_nt_bf16<<<dim3(ND3_ / 128, (int)(MT / 128)), blk, 0, stream>>>(
      x, w_qkv, qkv, nullptr, (int)MT, ND3_, DIM_);

  // 2) RoPE + split to bf16 [B,H,T,HD]
  rope_split<<<(B_ * T_ * H_ * 32) / 256, blk, 0, stream>>>(qkv, qb, kb2, vb2);

  // 3) flash attention -> attn [B*T, DIM] f32
  flash_attn<<<B_ * H_ * (T_ / 128), blk, 0, stream>>>(qb, kb2, vb2, attn);

  // 4) out = attn @ w_proj^T + b_proj   (M=8192, N=1024, K=1024)
  gemm_nt_bf16<<<dim3(DIM_ / 128, (int)(MT / 128)), blk, 0, stream>>>(
      attn, w_proj, out, b_proj, (int)MT, DIM_, DIM_);
}
